// Block_20753281974722
// MI455X (gfx1250) — matrix operand order repaired and checked by reading, not compiled, not run on hardware
//
#include <hip/hip_runtime.h>
#include <hip/hip_bf16.h>

// Problem constants (from reference setup_inputs)
constexpr int BB   = 16;
constexpr int HH   = 8;     // heads
constexpr int Hdim = 64;
constexpr int Wdim = 64;
constexpr int L    = Hdim * Wdim;   // 4096
constexpr int D    = 32;
constexpr int S    = 64;
constexpr int LTILE = 128;

typedef __attribute__((ext_vector_type(16))) _Float16 v16h;
typedef __attribute__((ext_vector_type(8)))  _Float16 v8h;
typedef __attribute__((ext_vector_type(8)))  float    v8f;

// ds_swizzle_b32 group-of-32 XOR patterns: offset[14:10]=xor, [9:5]=or, [4:0]=and
#define SWZ_X1  0x041F
#define SWZ_X2  0x081F
#define SWZ_X4  0x101F
#define SWZ_X8  0x201F
#define SWZ_X16 0x401F

#define SWZF(x, pat) __int_as_float(__builtin_amdgcn_ds_swizzle(__float_as_int(x), (pat)))
#define SWZI(x, pat) __builtin_amdgcn_ds_swizzle((x), (pat))

// ---------------------------------------------------------------------------
// Kernel 1: recover integer cluster assignment from hard one-hot [B,S,H,W]
// ---------------------------------------------------------------------------
__global__ __launch_bounds__(256) void k_idx(const float* __restrict__ delta,
                                             int* __restrict__ idx) {
  int t = blockIdx.x * 256 + threadIdx.x;          // b*L + l
  if (t >= BB * L) return;
  int b = t / L, l = t - b * L;
  const float* p = delta + (size_t)b * S * L + l;
  int best = 0;
  #pragma unroll 4
  for (int s = 0; s < S; ++s) {
    if (p[(size_t)s * L] > 0.5f) { best = s; break; }
  }
  idx[t] = best;
}

// ---------------------------------------------------------------------------
// Kernel 2: initial centers c0 = l2norm(scatter-sum xf by idx), stored as f16
// One block per (b,h). LDS accumulate, wave-per-pixel, lane = d index.
// ---------------------------------------------------------------------------
__global__ __launch_bounds__(256) void k_c0(const float* __restrict__ xf,
                                            const int* __restrict__ idx,
                                            _Float16* __restrict__ c0h) {
  __shared__ float csum[S * D];
  const int bh = blockIdx.x;
  const int b  = bh / HH;
  const int tid = threadIdx.x, lane = tid & 31, wave = tid >> 5;

  for (int i = tid; i < S * D; i += 256) csum[i] = 0.f;
  __syncthreads();

  const float* xs = xf + (size_t)bh * L * D;
  const int*   ib = idx + (size_t)b * L;
  for (int p = wave; p < L; p += 8) {
    int s = ib[p];
    float v = xs[(size_t)p * D + lane];
    atomicAdd(&csum[s * D + lane], v);
  }
  __syncthreads();

  for (int s = wave; s < S; s += 8) {
    float v = csum[s * D + lane];
    float sq = v * v;
    sq += SWZF(sq, SWZ_X1);
    sq += SWZF(sq, SWZ_X2);
    sq += SWZF(sq, SWZ_X4);
    sq += SWZF(sq, SWZ_X8);
    sq += SWZF(sq, SWZ_X16);
    float n = fmaxf(sqrtf(sq), 1e-12f);
    c0h[(size_t)bh * S * D + s * D + lane] = (_Float16)(v / n);
  }
}

// ---------------------------------------------------------------------------
// Kernel 3 (the WMMA core): per (b,h) x 128-pixel tile
//   affinity = xf_tile(128x32,f16) . c0^T(32x64,f16)  via v_wmma_f32_16x16x32_f16
//   -> branchless per-row argmax (ds_swizzle max/min reductions)
//   -> write onehot rows, LDS-accumulate c_sum and center coordinate sums
// ---------------------------------------------------------------------------
__global__ __launch_bounds__(256) void k_affinity(const float* __restrict__ xf,
                                                  const _Float16* __restrict__ c0h,
                                                  float* __restrict__ onehot_out,
                                                  float* __restrict__ csum_g,
                                                  float* __restrict__ cent_g) {
  __shared__ float                  xf_lds[LTILE * D];   // 16 KB
  __shared__ __align__(32) _Float16 xh_lds[LTILE * D];   //  8 KB
  __shared__ __align__(32) _Float16 ch_lds[S * D];       //  4 KB
  __shared__ float                  csum_lds[S * D];     //  8 KB
  __shared__ float                  cent_lds[3 * S];
  __shared__ int                    sstar[LTILE];

  const int bh = blockIdx.y;
  const int l0 = blockIdx.x * LTILE;
  const int tid = threadIdx.x, lane = tid & 31, wave = tid >> 5;

  for (int i = tid; i < S * D; i += 256) csum_lds[i] = 0.f;
  for (int i = tid; i < 3 * S; i += 256) cent_lds[i] = 0.f;

  const float* xsrc = xf + ((size_t)bh * L + l0) * D;
  for (int i = tid; i < LTILE * D; i += 256) {
    float v = xsrc[i];
    xf_lds[i] = v;
    xh_lds[i] = (_Float16)v;
  }
  const _Float16* csrc = c0h + (size_t)bh * S * D;
  for (int i = tid; i < S * D; i += 256) ch_lds[i] = csrc[i];
  __syncthreads();

  // --- WMMA: wave handles rows [wave*16, wave*16+16), all 64 clusters ---
  const int hi = lane >> 4;         // half-wave select
  const int lm = lane & 15;         // M (A) / N (B,C) within tile
  const int rowbase = wave * 16;

  // A fragment: 16x32 f16; lane lm = row; halves K = hi*16 + e
  v16h afrag = *(const v16h*)&xh_lds[(rowbase + lm) * D + hi * 16];

  // Issue all 4 WMMAs back-to-back (16 rows x 64 clusters, K=32 in one shot each)
  v8f acc[4];
  #pragma unroll
  for (int t = 0; t < 4; ++t) {
    // B fragment: 32x16 f16, lane lm = column s = t*16+lm; halves K = hi*16 + e
    v16h bfrag = *(const v16h*)&ch_lds[(t * 16 + lm) * D + hi * 16];
    v8f c = {};
    acc[t] = __builtin_amdgcn_wmma_f32_16x16x32_f16(
        /*neg_a=*/false, afrag, /*neg_b=*/false, bfrag,
        /*c_mod=*/(short)0, c, /*reuse_a=*/false, /*reuse_b=*/false);
  }

  // Branchless per-row argmax. Row m = r + hi*8; this lane holds column t*16+lm.
  #pragma unroll
  for (int r = 0; r < 8; ++r) {
    float v0 = acc[0][r], v1 = acc[1][r], v2 = acc[2][r], v3 = acc[3][r];
    // phase 1: row maximum
    float m = fmaxf(fmaxf(v0, v1), fmaxf(v2, v3));
    m = fmaxf(m, SWZF(m, SWZ_X1));
    m = fmaxf(m, SWZF(m, SWZ_X2));
    m = fmaxf(m, SWZF(m, SWZ_X4));
    m = fmaxf(m, SWZF(m, SWZ_X8));
    // phase 2: lowest column index attaining the maximum (exact fp equality)
    int c0i = (v0 == m) ? (0 * 16 + lm) : 0x7fffffff;
    int c1i = (v1 == m) ? (1 * 16 + lm) : 0x7fffffff;
    int c2i = (v2 == m) ? (2 * 16 + lm) : 0x7fffffff;
    int c3i = (v3 == m) ? (3 * 16 + lm) : 0x7fffffff;
    int cand = min(min(c0i, c1i), min(c2i, c3i));
    cand = min(cand, SWZI(cand, SWZ_X1));
    cand = min(cand, SWZI(cand, SWZ_X2));
    cand = min(cand, SWZI(cand, SWZ_X4));
    cand = min(cand, SWZI(cand, SWZ_X8));
    if (lm == 0) sstar[rowbase + hi * 8 + r] = cand;
  }
  __syncthreads();

  // --- scatter-accumulate c_sum and center coordinate sums (LDS) ---
  for (int r = 0; r < 16; ++r) {
    int row = rowbase + r;
    int s = sstar[row];
    atomicAdd(&csum_lds[s * D + lane], xf_lds[row * D + lane]);
    if (lane == 0) {
      int l = l0 + row;
      atomicAdd(&cent_lds[s], 1.0f);
      atomicAdd(&cent_lds[S + s], (float)(l >> 6));       // i = l / W
      atomicAdd(&cent_lds[2 * S + s], (float)(l & 63));   // j = l % W
    }
  }
  __syncthreads();

  // --- write onehot rows (contiguous LTILE*S block), float4 stores ---
  float4* orow = (float4*)(onehot_out + ((size_t)bh * L + l0) * S);
  for (int i = tid; i < LTILE * (S / 4); i += 256) {
    int row = i >> 4;            // S/4 == 16
    int s4  = (i & 15) * 4;
    int st  = sstar[row];
    float4 v = make_float4(0.f, 0.f, 0.f, 0.f);
    if (st >= s4 && st < s4 + 4) ((float*)&v)[st - s4] = 1.0f;
    orow[i] = v;
  }

  // --- flush block-local sums to global ---
  float* cg = csum_g + (size_t)bh * S * D;
  for (int i = tid; i < S * D; i += 256) atomicAdd(&cg[i], csum_lds[i]);
  float* tg = cent_g + (size_t)bh * 3 * S;
  for (int i = tid; i < 3 * S; i += 256) atomicAdd(&tg[i], cent_lds[i]);
}

// ---------------------------------------------------------------------------
// Kernel 4: c = l2norm(c_sum); centers = sums/size (0 or <1e-12 -> inf); scale
// One wave per (bh, s), lane = d index.
// ---------------------------------------------------------------------------
__global__ __launch_bounds__(256) void k_finalize(const float* __restrict__ csum_g,
                                                  const float* __restrict__ cent_g,
                                                  float* __restrict__ c_out,
                                                  float* __restrict__ centers,
                                                  const float* __restrict__ scale_param,
                                                  float* __restrict__ scale_out) {
  if (blockIdx.x == 0 && threadIdx.x == 0) *scale_out = 15.0f * scale_param[0];
  int gw = (blockIdx.x * 256 + threadIdx.x) >> 5;   // bh*S + s
  int lane = threadIdx.x & 31;
  if (gw >= BB * HH * S) return;

  float v = csum_g[(size_t)gw * D + lane];
  float sq = v * v;
  sq += SWZF(sq, SWZ_X1);
  sq += SWZF(sq, SWZ_X2);
  sq += SWZF(sq, SWZ_X4);
  sq += SWZF(sq, SWZ_X8);
  sq += SWZF(sq, SWZ_X16);
  float n = fmaxf(sqrtf(sq), 1e-12f);
  c_out[(size_t)gw * D + lane] = v / n;

  if (lane == 0) {
    int bh = gw / S, s = gw - bh * S;
    const float* tg = cent_g + (size_t)bh * 3 * S;
    float cnt = tg[s];
    if (cnt == 0.f) cnt = 1.f;
    float cy = tg[S + s] / cnt;
    float cx = tg[2 * S + s] / cnt;
    if (cy < 1e-12f) cy = __builtin_inff();
    if (cx < 1e-12f) cx = __builtin_inff();
    centers[(size_t)gw * 2]     = cy;
    centers[(size_t)gw * 2 + 1] = cx;
  }
}

// ---------------------------------------------------------------------------
// Kernel 5: rpb[b,h,l,s] = exp(-sqrt((i-cy)^2 + (j-cx)^2)); float4 per thread
// ---------------------------------------------------------------------------
__global__ __launch_bounds__(256) void k_rpb(const float* __restrict__ centers,
                                             float* __restrict__ rpb) {
  size_t t = (size_t)blockIdx.x * 256 + threadIdx.x;  // over (bh, l, s/4)
  int    s4 = (int)(t & (S / 4 - 1)) * 4;
  size_t r  = t >> 4;                                  // bh*L + l
  int l  = (int)(r & (L - 1));
  int bh = (int)(r >> 12);
  float fi = (float)(l >> 6);
  float fj = (float)(l & 63);
  const float* cb = centers + ((size_t)bh * S + s4) * 2;
  float4 out;
  #pragma unroll
  for (int k = 0; k < 4; ++k) {
    float dy = fi - cb[2 * k];
    float dx = fj - cb[2 * k + 1];
    float dist = sqrtf(dy * dy + dx * dx);             // inf center -> exp(-inf)=0
    ((float*)&out)[k] = __expf(-dist);
  }
  ((float4*)rpb)[t] = out;
}

// ---------------------------------------------------------------------------
extern "C" void kernel_launch(void* const* d_in, const int* in_sizes, int n_in,
                              void* d_out, int out_size, void* d_ws, size_t ws_size,
                              hipStream_t stream) {
  (void)in_sizes; (void)n_in; (void)out_size; (void)ws_size;
  const float* x           = (const float*)d_in[0];   // [B,h,H,W,d] fp32
  const float* delta       = (const float*)d_in[1];   // [B,S,H,W]  fp32
  const float* scale_param = (const float*)d_in[2];   // scalar

  // workspace layout
  char*     ws   = (char*)d_ws;
  int*      idx  = (int*)(ws + 0);                        // B*L ints      (256 KB)
  _Float16* c0h  = (_Float16*)(ws + 262144);              // B*h*S*D f16   (512 KB)
  float*    csum = (float*)(ws + 786432);                 // B*h*S*D f32   (  1 MB)
  float*    cent = (float*)(ws + 1835008);                // B*h*3*S f32   ( 96 KB)
  float*    ctr  = (float*)(ws + 2228224);                // B*h*S*2 f32   ( 64 KB)

  // output layout: onehot | c | rpb | scale  (flat concat, fp32)
  float* out       = (float*)d_out;
  float* onehot    = out;
  float* c_out     = onehot + (size_t)BB * HH * L * S;
  float* rpb       = c_out + (size_t)BB * HH * S * D;
  float* scale_out = rpb + (size_t)BB * HH * L * S;

  hipMemsetAsync(csum, 0, (size_t)BB * HH * S * D * sizeof(float), stream);
  hipMemsetAsync(cent, 0, (size_t)BB * HH * S * 3 * sizeof(float), stream);

  k_idx<<<(BB * L) / 256, 256, 0, stream>>>(delta, idx);
  k_c0<<<BB * HH, 256, 0, stream>>>(x, idx, c0h);

  dim3 grid_aff(L / LTILE, BB * HH);
  k_affinity<<<grid_aff, 256, 0, stream>>>(x, c0h, onehot, csum, cent);

  k_finalize<<<(BB * HH * S * 32) / 256, 256, 0, stream>>>(csum, cent, c_out, ctr,
                                                           scale_param, scale_out);

  k_rpb<<<(int)(((size_t)BB * HH * L * (S / 4)) / 256), 256, 0, stream>>>(ctr, rpb);
}
